// Attention_13554916786727
// MI455X (gfx1250) — compile-verified, matrix operand order checked by reading, not run on hardware
//
#include <hip/hip_runtime.h>

// Problem constants (B,H,S,D) = (4,16,4096,128)
#define BH     64
#define SEQ    4096
#define DIM    128
#define CHUNK  512
#define NSTEP  (CHUNK / 32)
#define NCHUNK (SEQ / CHUNK)
#define SCALE  0.25f   // 1/sqrt(n_heads=16)

typedef __attribute__((ext_vector_type(16))) __bf16        v16bf;
typedef __attribute__((ext_vector_type(8)))  float         v8f;
typedef __attribute__((ext_vector_type(4)))  unsigned int  v4u;
typedef __attribute__((ext_vector_type(8)))  int           v8i;
typedef __attribute__((ext_vector_type(4)))  int           v4i;

// K-index base for 16-bit WMMA A/B fragments (ISA 7.12.2):
// VGPR r in 0..3 -> K = {2r, 2r+1} (lanes 0-15), +8 (lanes 16-31)
// VGPR r in 4..7 -> K = 16 + {2(r-4), ...}, +8 for upper lane half
__device__ __forceinline__ int kbase(int r, int g) {
    return ((r & 4) << 2) + (g << 3) + ((r & 3) << 1);
}

// ---------------------------------------------------------------------------
// Tensor Data Mover: DMA a 2-D f32 tile [rows x 128] (row stride 128 elems)
// from global memory into LDS. D# layout per CDNA5 ISA ch.8:
//   group0: count=1 | lds_addr | global_addr[56:0] | type=2
//   group1: data_size=4B | tensor_dim0=128 | tensor_dim1 | tile_dim0=128 |
//           tile_dim1=rows | tensor_dim0_stride=128
// Groups 2/3 zero (<=2-D tensor). Tracked on TENSORcnt.
// ---------------------------------------------------------------------------
__device__ __forceinline__ void tdm_load_tile(const float* gptr, unsigned lds_off,
                                              int rows) {
    unsigned long long ga = (unsigned long long)(size_t)gptr;
    v4u g0;
    g0[0] = 1u;                                               // count=1, user D#
    g0[1] = lds_off;                                          // LDS byte address
    g0[2] = (unsigned)(ga & 0xFFFFFFFFu);                     // global_addr[31:0]
    g0[3] = (unsigned)((ga >> 32) & 0x01FFFFFFu) | (2u << 30);// [56:32] | type=2
    v8i g1;
    g1[0] = (int)(2u << 16);          // workgroup_mask=0, data_size=2 (4 bytes)
    g1[1] = (int)((unsigned)DIM << 16);   // tensor_dim0[15:0] = 128
    g1[2] = (int)((unsigned)SEQ << 16);   // tensor_dim0 hi=0, tensor_dim1 lo
    g1[3] = (int)((unsigned)DIM << 16);   // tensor_dim1 hi=0, tile_dim0 = 128
    g1[4] = rows;                          // tile_dim1 = rows, tile_dim2 = 0
    g1[5] = DIM;                           // tensor_dim0_stride = 128
    g1[6] = 0;
    g1[7] = 0;
    v4i z4 = (v4i)0;
#if __clang_major__ >= 23
    v8i z8 = (v8i)0;
    __builtin_amdgcn_tensor_load_to_lds(g0, g1, z4, z4, z8, 0);
#else
    __builtin_amdgcn_tensor_load_to_lds(g0, g1, z4, z4, 0);
#endif
}

// ---------------------------------------------------------------------------
// Kernel 1: partial score = Q^T K over an S-chunk, atomically accumulated.
// Block = 256 threads (8 waves). TDM double-buffers 32x128 f32 Q/K tiles into
// LDS; f32->bf16 conversion happens in fragment assembly (co-executes with
// the XDL WMMAs).
// ---------------------------------------------------------------------------
__global__ __launch_bounds__(256) void score_kernel(const float* __restrict__ q,
                                                    const float* __restrict__ k,
                                                    float* __restrict__ score) {
    __shared__ float Qf[2][32][DIM];   // 2 x 16 KiB
    __shared__ float Kf[2][32][DIM];   // 2 x 16 KiB

    const int bid   = blockIdx.x;
    const int bh    = bid / NCHUNK;
    const int s0    = (bid % NCHUNK) * CHUNK;
    const int tid   = threadIdx.x;
    const int wave  = tid >> 5;
    const int lane  = tid & 31;
    const int l15   = lane & 15;
    const int g     = lane >> 4;

    const float* qb = q + (size_t)bh * SEQ * DIM;
    const float* kb = k + (size_t)bh * SEQ * DIM;

    v8f acc[8];
    #pragma unroll
    for (int n = 0; n < 8; ++n) acc[n] = (v8f)0.0f;

    // prologue: DMA first Q/K tiles into buffer 0
    if (wave == 0) {
        tdm_load_tile(qb + (size_t)s0 * DIM, (unsigned)(size_t)&Qf[0][0][0], 32);
        tdm_load_tile(kb + (size_t)s0 * DIM, (unsigned)(size_t)&Kf[0][0][0], 32);
    }

    for (int i = 0; i < NSTEP; ++i) {
        const int buf = i & 1;
        if (wave == 0) {
            if (i + 1 < NSTEP) {
                const int ssn = s0 + 32 * (i + 1);
                const int nb  = (i + 1) & 1;
                tdm_load_tile(qb + (size_t)ssn * DIM,
                              (unsigned)(size_t)&Qf[nb][0][0], 32);
                tdm_load_tile(kb + (size_t)ssn * DIM,
                              (unsigned)(size_t)&Kf[nb][0][0], 32);
                // allow the 2 just-issued DMAs to stay in flight; the 2 for
                // this iteration (issued last time) must be complete
                __builtin_amdgcn_s_wait_tensorcnt((short)2);
            } else {
                __builtin_amdgcn_s_wait_tensorcnt((short)0);
            }
        }
        __syncthreads();   // publish TDM-written tile to all waves

        // ---- A fragment: rows d = 16*wave + l15 of Q^T (f32->bf16 cvt) ----
        const int d = (wave << 4) + l15;
        v16bf a;
        #pragma unroll
        for (int r = 0; r < 8; ++r) {
            const int kb2 = kbase(r, g);
            a[2 * r]     = (__bf16)Qf[buf][kb2][d];
            a[2 * r + 1] = (__bf16)Qf[buf][kb2 + 1][d];
        }

        // ---- 8 column tiles of K, WMMA accumulate ----
        #pragma unroll
        for (int n = 0; n < 8; ++n) {
            const int e = (n << 4) + l15;
            v16bf bb;
            #pragma unroll
            for (int r = 0; r < 8; ++r) {
                const int kb2 = kbase(r, g);
                bb[2 * r]     = (__bf16)Kf[buf][kb2][e];
                bb[2 * r + 1] = (__bf16)Kf[buf][kb2 + 1][e];
            }
            acc[n] = __builtin_amdgcn_wmma_f32_16x16x32_bf16(
                false, a, false, bb, (short)0, acc[n], false, false);
        }
        __syncthreads();   // reads done before this buffer is DMA'd again
    }

    // ---- accumulate partial scores (f32 atomics, L2-resident 4 MiB) ----
    float* sc = score + (size_t)bh * DIM * DIM;
    #pragma unroll
    for (int n = 0; n < 8; ++n) {
        const int e = (n << 4) + l15;
        #pragma unroll
        for (int i = 0; i < 8; ++i) {
            const int dr = (wave << 4) + (g << 3) + i;
            atomicAdd(&sc[dr * DIM + e], acc[n][i]);
        }
    }
}

// ---------------------------------------------------------------------------
// Kernel 2: row softmax of score (scaled by 1/sqrt(H)), emit bf16 probs.
// One wave32 per 128-wide row; 8 rows per 256-thread block.
// ---------------------------------------------------------------------------
__global__ __launch_bounds__(256) void softmax_kernel(const float* __restrict__ score,
                                                      __bf16* __restrict__ probs) {
    const int wave = threadIdx.x >> 5;
    const int lane = threadIdx.x & 31;
    const int row  = blockIdx.x * 8 + wave;          // 0 .. BH*DIM-1

    const float* sr = score + (size_t)row * DIM;
    float x[4];
    float mx = -3.402823466e38f;
    #pragma unroll
    for (int j = 0; j < 4; ++j) {
        x[j] = sr[lane + 32 * j] * SCALE;
        mx = fmaxf(mx, x[j]);
    }
    #pragma unroll
    for (int off = 16; off > 0; off >>= 1)
        mx = fmaxf(mx, __shfl_xor(mx, off, 32));

    float sum = 0.0f;
    #pragma unroll
    for (int j = 0; j < 4; ++j) {
        x[j] = __expf(x[j] - mx);
        sum += x[j];
    }
    #pragma unroll
    for (int off = 16; off > 0; off >>= 1)
        sum += __shfl_xor(sum, off, 32);

    const float inv = 1.0f / sum;
    __bf16* pr = probs + (size_t)row * DIM;
    #pragma unroll
    for (int j = 0; j < 4; ++j)
        pr[lane + 32 * j] = (__bf16)(x[j] * inv);
}

// ---------------------------------------------------------------------------
// Kernel 3: out[d][s] = sum_e P[d][e] * V[s][e]  (128 x 4096 per (b,h)).
// Block = 256 threads (8 waves) computes a 128(d) x 128(s) tile. The 64 KiB
// V tile is TDM-staged into LDS once per block (removes 8x per-wave global
// duplication); contraction K = 128 -> 4 bf16 WMMA k-steps per output tile.
// ---------------------------------------------------------------------------
__global__ __launch_bounds__(256) void out_kernel(const __bf16* __restrict__ probs,
                                                  const float* __restrict__ v,
                                                  float* __restrict__ out) {
    __shared__ float Vf[128][DIM];                   // 64 KiB: V[st+s][e]

    const int bh   = blockIdx.x >> 5;
    const int st   = (blockIdx.x & 31) * 128;        // s-tile base
    const int wave = threadIdx.x >> 5;
    const int lane = threadIdx.x & 31;
    const int l15  = lane & 15;
    const int g    = lane >> 4;

    const __bf16* pbase = probs + (size_t)bh * DIM * DIM;
    const float*  vbase = v + (size_t)bh * SEQ * DIM;

    if (wave == 0) {
        tdm_load_tile(vbase + (size_t)st * DIM, (unsigned)(size_t)&Vf[0][0], 128);
        __builtin_amdgcn_s_wait_tensorcnt((short)0);
    }
    __syncthreads();

    v8f acc[8];
    #pragma unroll
    for (int n = 0; n < 8; ++n) acc[n] = (v8f)0.0f;

    const int dl = (wave << 4) + l15;                // this lane's d-row for A

    #pragma unroll
    for (int kk = 0; kk < 4; ++kk) {                 // e-base = 32*kk
        const int eb = kk << 5;

        // A fragment from bf16 probabilities (row-major, dword loads, L2-hot)
        const __bf16* pr = pbase + (size_t)dl * DIM + eb;
        typedef __attribute__((ext_vector_type(8))) unsigned int v8u;
        v8u au;
        #pragma unroll
        for (int r = 0; r < 8; ++r)
            au[r] = *(const unsigned int*)&pr[kbase(r, g)];
        v16bf a = __builtin_bit_cast(v16bf, au);

        #pragma unroll
        for (int n = 0; n < 8; ++n) {
            const int srow = (n << 4) + l15;         // local s row in LDS tile
            v16bf bb;
            #pragma unroll
            for (int r = 0; r < 8; ++r) {
                float2 f = *(const float2*)&Vf[srow][eb + kbase(r, g)];
                bb[2 * r]     = (__bf16)f.x;
                bb[2 * r + 1] = (__bf16)f.y;
            }
            acc[n] = __builtin_amdgcn_wmma_f32_16x16x32_bf16(
                false, a, false, bb, (short)0, acc[n], false, false);
        }
    }

    // store f32 result: out[(bh)][d][s]
    float* ob = out + (size_t)bh * DIM * SEQ;
    #pragma unroll
    for (int n = 0; n < 8; ++n) {
        const int sc = st + (n << 4) + l15;
        #pragma unroll
        for (int i = 0; i < 8; ++i) {
            const int dr = (wave << 4) + (g << 3) + i;
            ob[(size_t)dr * SEQ + sc] = acc[n][i];
        }
    }
}

// ---------------------------------------------------------------------------
extern "C" void kernel_launch(void* const* d_in, const int* in_sizes, int n_in,
                              void* d_out, int out_size, void* d_ws, size_t ws_size,
                              hipStream_t stream) {
    const float* q = (const float*)d_in[0];
    const float* k = (const float*)d_in[1];
    const float* v = (const float*)d_in[2];
    float* out = (float*)d_out;

    // workspace layout: [0, 4 MiB) f32 score accumulators, then 2 MiB bf16 probs
    float*  score = (float*)d_ws;
    __bf16* probs = (__bf16*)((char*)d_ws + (size_t)BH * DIM * DIM * sizeof(float));

    hipMemsetAsync(score, 0, (size_t)BH * DIM * DIM * sizeof(float), stream);

    score_kernel<<<BH * NCHUNK, 256, 0, stream>>>(q, k, score);
    softmax_kernel<<<(BH * DIM) / 8, 256, 0, stream>>>(score, probs);
    out_kernel<<<BH * (SEQ / 128), 256, 0, stream>>>(probs, v, out);
}
